// VmfDiff_57423712748223
// MI455X (gfx1250) — matrix operand order repaired and checked by reading, not compiled, not run on hardware
//
#include <hip/hip_runtime.h>
#include <cmath>
#include <math.h>

#ifndef M_PI
#define M_PI 3.14159265358979323846
#endif

typedef __attribute__((ext_vector_type(16))) _Float16 v16h;
typedef __attribute__((ext_vector_type(8)))  float    v8f;

#define NT 7            // N tiles: 7*16 = 112 >= L (=100), zero padded
#define M_TERMS 256

// ---------------- RNG helpers (counter/hash based, deterministic) ----------------
__device__ __forceinline__ unsigned int mixu(unsigned int x) {
  x ^= x >> 16; x *= 0x7feb352dU;
  x ^= x >> 15; x *= 0x846ca68bU;
  x ^= x >> 16; return x;
}
__device__ __forceinline__ float u01(unsigned int h) {
  return ((h >> 8) + 0.5f) * (1.0f / 16777216.0f);
}
struct Rng {
  unsigned int s;
  __device__ __forceinline__ float uni() { s = s * 1664525u + 1013904223u; return u01(mixu(s)); }
  __device__ __forceinline__ float nrm() {
    float u1 = uni(), u2 = uni();
    return sqrtf(-2.0f * logf(u1)) * __cosf(6.28318530718f * u2);
  }
};
// deterministic normal indexed by (base, j): regenerable across passes
__device__ __forceinline__ float hnrm(unsigned int base, int j) {
  unsigned int h1 = mixu(base + (unsigned int)(2 * j) + 1u);
  unsigned int h2 = mixu(base + (unsigned int)(2 * j) + 2u);
  return sqrtf(-2.0f * logf(u01(h1))) * __cosf(6.28318530718f * u01(h2));
}
// Marsaglia-Tsang gamma(a), a > 1
__device__ __forceinline__ float gamma_mt(Rng& r, float a) {
  float d = a - (1.0f / 3.0f);
  float c = rsqrtf(9.0f * d);
  for (int i = 0; i < 32; ++i) {
    float x = r.nrm();
    float t = 1.0f + c * x;
    if (t <= 0.0f) continue;
    float v = t * t * t;
    float u = r.uni();
    if (logf(u) < 0.5f * x * x + d - d * v + d * logf(v)) return d * v;
  }
  return d;
}

// ---------------- Kernel 1: pre-swizzle W_mu (f32 row-major) into WMMA B layout (f16) ----------
// B tile layout (16-bit, 32x16): lane = 16*((K>>3)&1) + N ; dword v holds K = 8*(lane>>4)+2v+8*(v>>2), +1.
// Flat uint index: ((kk*NT + t)*32 + lane)*8 + v
__global__ void pack_w_f16(const float* __restrict__ Wmu, unsigned int* __restrict__ Bpk,
                           int H, int L, int total) {
  int id = blockIdx.x * blockDim.x + threadIdx.x;
  if (id >= total) return;
  int v = id & 7;
  int rest = id >> 3;
  int lane = rest & 31;
  rest >>= 5;
  int t  = rest % NT;
  int kk = rest / NT;
  int K0 = kk * 32 + ((lane >> 4) << 3) + 2 * v + ((v >> 2) << 3);
  int n  = t * 16 + (lane & 15);
  float w0 = (n < L) ? Wmu[(size_t)K0 * L + n]       : 0.0f;
  float w1 = (n < L) ? Wmu[(size_t)(K0 + 1) * L + n] : 0.0f;
  union { unsigned int u; _Float16 h[2]; } p;
  p.h[0] = (_Float16)w0;
  p.h[1] = (_Float16)w1;
  Bpk[id] = p.u;
}

// ---------------- Kernel 2: fused WMMA GEMM (mu) + GEMV (kappa) + norm epilogue -------------
// Block = 256 threads = 8 waves; wave w owns rows r0 = blockIdx*128 + w*16; full N strip (7 tiles).
__global__ void __launch_bounds__(256)
vmf_gemm_wmma(const float* __restrict__ lat, const unsigned int* __restrict__ Bpk,
              const float* __restrict__ bmu, const float* __restrict__ Wk,
              const float* __restrict__ bk,
              float* __restrict__ mu_out, float* __restrict__ kap_out,
              float* __restrict__ red_out,
              int H, int L, int B) {
  __shared__ float s_wk[2048];        // W_kappa (H <= 2048)
  __shared__ float s_bmu[NT * 16];    // b_mu, zero padded to 112
  for (int i = threadIdx.x; i < H; i += blockDim.x) s_wk[i] = Wk[i];
  for (int i = threadIdx.x; i < NT * 16; i += blockDim.x) s_bmu[i] = (i < L) ? bmu[i] : 0.0f;
  __syncthreads();

  const int lane = threadIdx.x & 31;
  const int wv   = threadIdx.x >> 5;
  const int r0   = blockIdx.x * 128 + wv * 16;
  if (r0 >= B) return;
  const int mrow = r0 + (lane & 15);          // A layout: lane l holds row M = l&15
  const int kg   = (lane >> 4) << 3;          // K sub-group: 0 (lanes 0-15) or 8 (lanes 16-31)
  const float* arow = lat + (size_t)mrow * H;

  v8f acc[NT];
  v8f zero = {0.f, 0.f, 0.f, 0.f, 0.f, 0.f, 0.f, 0.f};
#pragma unroll
  for (int t = 0; t < NT; ++t) acc[t] = zero;
  float kacc = 0.0f;

  const int ksteps = H >> 5;
  for (int kk = 0; kk < ksteps; ++kk) {
    const int k0 = kk << 5;
    // A tile: 16 fp32 per lane in the exact 16x32 f16 A layout
    float4 a0 = *(const float4*)(arow + k0 + kg);
    float4 a1 = *(const float4*)(arow + k0 + kg + 4);
    float4 a2 = *(const float4*)(arow + k0 + 16 + kg);
    float4 a3 = *(const float4*)(arow + k0 + 16 + kg + 4);
    if (k0 + 32 < H) __builtin_prefetch(arow + k0 + 32 + kg, 0, 1);

    // fused kappa GEMV partial on the fp32 values (before f16 conversion)
    const float* wkp = s_wk + k0 + kg;
    const float* wkq = s_wk + k0 + 16 + kg;
    kacc += a0.x * wkp[0] + a0.y * wkp[1] + a0.z * wkp[2] + a0.w * wkp[3]
          + a1.x * wkp[4] + a1.y * wkp[5] + a1.z * wkp[6] + a1.w * wkp[7]
          + a2.x * wkq[0] + a2.y * wkq[1] + a2.z * wkq[2] + a2.w * wkq[3]
          + a3.x * wkq[4] + a3.y * wkq[5] + a3.z * wkq[6] + a3.w * wkq[7];

    union { v16h v; _Float16 h[16]; } A;
    A.h[0]  = (_Float16)a0.x; A.h[1]  = (_Float16)a0.y; A.h[2]  = (_Float16)a0.z; A.h[3]  = (_Float16)a0.w;
    A.h[4]  = (_Float16)a1.x; A.h[5]  = (_Float16)a1.y; A.h[6]  = (_Float16)a1.z; A.h[7]  = (_Float16)a1.w;
    A.h[8]  = (_Float16)a2.x; A.h[9]  = (_Float16)a2.y; A.h[10] = (_Float16)a2.z; A.h[11] = (_Float16)a2.w;
    A.h[12] = (_Float16)a3.x; A.h[13] = (_Float16)a3.y; A.h[14] = (_Float16)a3.z; A.h[15] = (_Float16)a3.w;

    const uint4* bp = (const uint4*)Bpk + ((size_t)kk * NT * 32 + lane) * 2;
#pragma unroll
    for (int t = 0; t < NT; ++t) {
      union { uint4 q[2]; v16h v; } Bm;
      const uint4* b = bp + (size_t)t * 64;
      Bm.q[0] = b[0];
      Bm.q[1] = b[1];
      acc[t] = __builtin_amdgcn_wmma_f32_16x16x32_f16(
          /*neg_a=*/false, A.v, /*neg_b=*/false, Bm.v,
          /*c_mod=*/(short)0, acc[t], /*reuse_a=*/false, /*reuse_b=*/false);
    }
  }

  // ---- kappa epilogue: combine complementary K halves (lane l <-> l+16) ----
  float kdot = kacc + __shfl_xor(kacc, 16, 32);
  if (lane < 16) {
    float kv = (kdot + bk[0]) * 10.0f + 50.0f;
    kv = fminf(fmaxf(kv, 10.0f), 150.0f);
    kap_out[r0 + lane] = kv;
  }

  // ---- bias add + row norms. C layout: VGPR v -> row v + 8*(lane>>4), lane&15 -> column ----
  const int nh = lane & 15;
  const int rb = r0 + ((lane >> 4) << 3);
  float ss[8];
#pragma unroll
  for (int v = 0; v < 8; ++v) ss[v] = 0.0f;
#pragma unroll
  for (int t = 0; t < NT; ++t) {
    int n = nh + (t << 4);
    float b = s_bmu[n];                 // 0 in padded columns; acc there is also 0
#pragma unroll
    for (int v = 0; v < 8; ++v) {
      float val = acc[t][v] + b;
      acc[t][v] = val;
      ss[v] += val * val;
    }
  }
#pragma unroll
  for (int d = 1; d < 16; d <<= 1) {
#pragma unroll
    for (int v = 0; v < 8; ++v) ss[v] += __shfl_xor(ss[v], d, 32);
  }
  float rn[8];
#pragma unroll
  for (int v = 0; v < 8; ++v) {
    float nm = sqrtf(ss[v]);
    rn[v] = 1.0f / nm;
    if (nh == 0) red_out[rb + v] = (nm - 1.0f) * (nm - 1.0f);
  }
#pragma unroll
  for (int t = 0; t < NT; ++t) {
    int n = nh + (t << 4);
    if (n < L) {
#pragma unroll
      for (int v = 0; v < 8; ++v)
        mu_out[(size_t)(rb + v) * L + n] = acc[t][v] * rn[v];
    }
  }
}

// ---------------- Kernel 3: Bessel-series KLD + Wood rejection vMF sampling ------------------
__global__ void __launch_bounds__(256)
vmf_kld_sample(const float* __restrict__ mu, const float* __restrict__ kap,
               float* __restrict__ kld_out, float* __restrict__ vec_out,
               int B, int L, int n_sample, float vh, float kconst) {
  __shared__ float c1[M_TERMS], c2[M_TERMS];  // lgamma(m+1)+lgamma(v+m+1), x-independent
  {
    int m = threadIdx.x;  // blockDim == M_TERMS
    float mf = (float)m;
    float lgm = lgammaf(mf + 1.0f);
    c1[m] = lgm + lgammaf(vh + mf + 1.0f);
    c2[m] = lgm + lgammaf(vh + mf);        // v = vh-1  ->  lgamma(v+m+1) = lgamma(vh+m)
  }
  __syncthreads();
  int row = blockIdx.x * blockDim.x + threadIdx.x;
  if (row >= B) return;

  float k  = kap[row];
  float lt = logf(0.5f * k);
  // pass 1: running maxima for stable logsumexp
  float mx1 = -1e30f, mx2 = -1e30f;
  for (int m = 0; m < M_TERMS; ++m) {
    float fm2 = 2.0f * (float)m;
    float l1 = (vh + fm2) * lt - c1[m];
    float l2 = (vh - 1.0f + fm2) * lt - c2[m];
    mx1 = fmaxf(mx1, l1);
    mx2 = fmaxf(mx2, l2);
  }
  // pass 2: sums
  float s1 = 0.0f, s2 = 0.0f;
  for (int m = 0; m < M_TERMS; ++m) {
    float fm2 = 2.0f * (float)m;
    float l1 = (vh + fm2) * lt - c1[m];
    float l2 = (vh - 1.0f + fm2) * lt - c2[m];
    s1 += __expf(l1 - mx1);
    s2 += __expf(l2 - mx2);
  }
  float li_h   = mx1 + logf(s1);
  float li_hm1 = mx2 + logf(s2);
  kld_out[row] = k * __expf(li_h - li_hm1) + (vh - 1.0f) * logf(k) - li_hm1 + kconst;

  // Wood (1994) rejection sampler for radial weight w
  float mdim = (float)(L - 1);
  float bb = mdim / (sqrtf(4.0f * k * k + mdim * mdim) + 2.0f * k);
  float x0 = (1.0f - bb) / (1.0f + bb);
  float cW = k * x0 + mdim * logf(1.0f - x0 * x0);
  float a  = 0.5f * mdim;  // beta(a, a)

  const float* mrow = mu + (size_t)row * L;
  for (int n = 0; n < n_sample; ++n) {
    Rng r;
    r.s = mixu((unsigned int)row * 0x9E3779B9u ^ (unsigned int)(n + 1) * 0x85EBCA6Bu);
    float w = x0;
    for (int it = 0; it < 64; ++it) {
      float g1 = gamma_mt(r, a), g2 = gamma_mt(r, a);
      float z  = g1 / (g1 + g2);
      float wp = (1.0f - (1.0f + bb) * z) / (1.0f - (1.0f - bb) * z);
      float u  = r.uni();
      w = wp;
      if (k * wp + mdim * logf(1.0f - x0 * wp) - cW >= logf(u)) break;
    }
    // tangent direction: Gram-Schmidt vs mu, three passes with regenerable normals
    unsigned int base = mixu((unsigned int)row ^ (unsigned int)(n * 0x27d4eb2d) ^ 0xA5A5A5A5u);
    float dot = 0.0f;
    for (int j = 0; j < L; ++j) dot += mrow[j] * hnrm(base, j);
    float ssq = 0.0f;
    for (int j = 0; j < L; ++j) { float o = hnrm(base, j) - mrow[j] * dot; ssq += o * o; }
    float inv = rsqrtf(ssq);
    float sf  = sqrtf(fmaxf(0.0f, 1.0f - w * w));
    float* vo = vec_out + ((size_t)n * B + row) * L;
    for (int j = 0; j < L; ++j) {
      float o = hnrm(base, j) - mrow[j] * dot;
      vo[j] = o * inv * sf + mrow[j] * w;
    }
  }
}

// ---------------- Host launcher ----------------
extern "C" void kernel_launch(void* const* d_in, const int* in_sizes, int n_in,
                              void* d_out, int out_size, void* d_ws, size_t ws_size,
                              hipStream_t stream) {
  const float* lat = (const float*)d_in[0];
  const float* Wmu = (const float*)d_in[1];
  const float* bmu = (const float*)d_in[2];
  const float* Wk  = (const float*)d_in[3];
  const float* bk  = (const float*)d_in[4];

  int L = in_sizes[2];
  int H = in_sizes[1] / L;
  int B = in_sizes[0] / H;

  float* out     = (float*)d_out;
  float* mu_out  = out;
  float* kap_out = out + (size_t)B * L;
  float* red_out = kap_out + B;
  float* kld_out = red_out + B;
  float* vec_out = kld_out + B;
  long long vecn = (long long)out_size - ((long long)B * L + 3LL * B);
  int n_sample = (int)(vecn / ((long long)B * L));
  if (n_sample < 1) n_sample = 1;

  // 1) swizzle W_mu into WMMA B layout (f16) in workspace
  unsigned int* Bpk = (unsigned int*)d_ws;
  int ksteps = H / 32;
  int total  = ksteps * NT * 32 * 8;
  pack_w_f16<<<(total + 255) / 256, 256, 0, stream>>>(Wmu, Bpk, H, L, total);

  // 2) fused WMMA GEMM + kappa GEMV + norm epilogue
  int blocks = (B + 127) / 128;
  vmf_gemm_wmma<<<blocks, 256, 0, stream>>>(lat, Bpk, bmu, Wk, bk,
                                            mu_out, kap_out, red_out, H, L, B);

  // 3) KLD + sampling
  double d  = (double)L;
  double kc = log(M_PI) * d / 2.0 + log(2.0) - lgamma(d / 2.0) - d / 2.0 * log(2.0 * M_PI);
  vmf_kld_sample<<<(B + 255) / 256, 256, 0, stream>>>(mu_out, kap_out, kld_out, vec_out,
                                                      B, L, n_sample,
                                                      (float)(d / 2.0), (float)kc);
}
